// GenerativeODE_19155554140946
// MI455X (gfx1250) — matrix operand order repaired and checked by reading, not compiled, not run on hardware
//
#include <hip/hip_runtime.h>
#include <hip/hip_bf16.h>
#include <math.h>

// ---------------- model constants ----------------
#define BATCH 1024
#define PF    16
#define NZ    2000
#define ZED   64
#define PED   32
#define NP    8
#define HID   256
#define ENC   512
#define OHID  512
#define NRB   3
#define TT    48
#define STATE 352            // HID + ZED + PED

// ---------------- WMMA plumbing (CDNA5 / gfx1250, wave32) ----------------
typedef __bf16 v16bf __attribute__((ext_vector_type(16)));
typedef float  v8f   __attribute__((ext_vector_type(8)));

union ABfrag { v16bf v; uint4 q[2]; unsigned short u[16]; };
union CFrag  { v8f   v; float f[8]; };

__device__ __forceinline__ unsigned short f32_to_bf16_bits(float f) {
  unsigned u = __float_as_uint(f);
  u += 0x7FFFu + ((u >> 16) & 1u);          // round-to-nearest-even
  return (unsigned short)(u >> 16);
}

// A-matrix 16x32 bf16 fragment, per ISA layout:
//  lanes 0-15: M=lane,     elems 0-7 -> K=0..7,   elems 8-15 -> K=16..23
//  lanes16-31: M=lane-16,  elems 0-7 -> K=8..15,  elems 8-15 -> K=24..31
// Row-major A (stride in elements, multiple of 8) => two 16B loads per frag.
__device__ __forceinline__ ABfrag load_a_frag(const unsigned short* A, int strideA,
                                              int kt, int lcol, int hi) {
  ABfrag a;
  const unsigned short* ap = A + (size_t)lcol * strideA + kt * 32 + 8 * hi;
  a.q[0] = *(const uint4*)(ap);
  a.q[1] = *(const uint4*)(ap + 16);
  return a;
}

// B-matrix 32x16 bf16 fragment, packed so lane's 16 elems are contiguous:
//  lane L -> N = L&15, K = (e&7) + 8*(L>=16) + 16*(e>>3). One tile = 32 lanes * 32B = 1KB.
__device__ __forceinline__ ABfrag load_b_frag(const unsigned short* Bpk,
                                              size_t tileIdx, int lane) {
  ABfrag b;
  const uint4* bp = (const uint4*)(Bpk + ((tileIdx * 32 + (size_t)lane) << 4));
  b.q[0] = bp[0];
  b.q[1] = bp[1];
  return b;
}

__device__ __forceinline__ v8f wmma_bf16(const ABfrag& a, const ABfrag& b, v8f c) {
  return __builtin_amdgcn_wmma_f32_16x16x32_bf16(false, a.v, false, b.v,
                                                 (short)0, c, false, false);
}

// ---------------- weight packer: f32 row-major (KxN) -> bf16 B-fragment tiles
// tile index = nt*Ktiles + kt; one 32-thread wave per tile.
__global__ __launch_bounds__(32)
void pack_b_kernel(const float* __restrict__ W, unsigned short* __restrict__ P,
                   int K, int N, int Ktiles, int Ntiles) {
  int t    = blockIdx.x;
  int kt   = t % Ktiles;
  int nt   = t / Ktiles;
  int lane = threadIdx.x & 31;
  int hi   = lane >> 4;
  int lcol = lane & 15;
  int n = nt * 16 + lcol;
  unsigned short* dst = P + (((size_t)t * 32 + lane) << 4);
#pragma unroll
  for (int e = 0; e < 16; ++e) {
    int k = kt * 32 + (e & 7) + 8 * hi + 16 * (e >> 3);
    float v = (k < K && n < N) ? W[(size_t)k * N + n] : 0.0f;
    dst[e] = f32_to_bf16_bits(v);
  }
}

// ---------------- generic WMMA GEMM: C = act(A(bf16,MxK) * Bpk + bias) ------
// blockDim = 128 (4 waves). grid = (M/16, ceil(Ntiles/4)). Each wave: 1 n-tile.
__global__ __launch_bounds__(128)
void gemm_kernel(const unsigned short* __restrict__ A, int strideA,
                 const unsigned short* __restrict__ Bpk,
                 const float* __restrict__ bias,
                 int Ktiles, int Ntiles, int act,
                 float* __restrict__ Cf, unsigned short* __restrict__ Cb,
                 int ldC, int Nreal) {
  int lane = threadIdx.x & 31;
  int wave = threadIdx.x >> 5;
  int hi   = lane >> 4;
  int lcol = lane & 15;
  int nt = blockIdx.y * 4 + wave;
  if (nt >= Ntiles) return;
  int m0 = blockIdx.x * 16;
  const unsigned short* Arow = A + (size_t)m0 * strideA;

  CFrag c;
  int ncol = nt * 16 + lcol;
  float bv = (ncol < Nreal) ? bias[ncol] : 0.0f;
#pragma unroll
  for (int r = 0; r < 8; ++r) c.f[r] = bv;

  size_t tbase = (size_t)nt * Ktiles;
  for (int kt = 0; kt < Ktiles; ++kt) {
    ABfrag a = load_a_frag(Arow, strideA, kt, lcol, hi);
    ABfrag b = load_b_frag(Bpk, tbase + kt, lane);
    __builtin_prefetch((const char*)Bpk + ((tbase + kt + 1) << 10), 0, 3);
    c.v = wmma_bf16(a, b, c.v);
  }

  if (ncol < Nreal) {
#pragma unroll
    for (int r = 0; r < 8; ++r) {
      int m = m0 + r + 8 * hi;
      float v = c.f[r];
      if (act == 1) v = fmaxf(v, 0.0f);
      if (Cf) Cf[(size_t)m * ldC + ncol] = v;
      if (Cb) Cb[(size_t)m * ldC + ncol] = f32_to_bf16_bits(v);
    }
  }
}

// ---------------- in-workgroup matmuls for the ODE (A in LDS) ---------------
// modes: 0 = relu -> f32out(ldF) + bfout   (W_in)
//        1 = tanh -> bfout only            (rb first matmul)
//        2 = tanh(f32out + acc) -> f32out + bfout  (rb second matmul, residual)
//        3 = none -> f32out(ldF)           (W_out -> k)
template <int MODE>
__device__ __forceinline__
void ode_epilogue(float v, int m, int n, float* f32out, int ldF,
                  unsigned short* bfout) {
  if (MODE == 0) {
    v = fmaxf(v, 0.0f);
    f32out[m * ldF + n] = v;
    bfout[m * 512 + n]  = f32_to_bf16_bits(v);
  } else if (MODE == 1) {
    v = tanhf(v);
    bfout[m * 512 + n] = f32_to_bf16_bits(v);
  } else if (MODE == 2) {
    v = tanhf(f32out[m * ldF + n] + v);
    f32out[m * ldF + n] = v;
    bfout[m * 512 + n]  = f32_to_bf16_bits(v);
  } else {
    f32out[m * ldF + n] = v;
  }
}

// Ntiles == 32 fast path: 8 waves x 4 n-tiles, K-outer, A fragment loaded ONCE
// per K step and fed to 4 independent wmma chains (A-reuse, 4x fewer ds loads).
template <int MODE>
__device__ __forceinline__
void wg_matmul32(const unsigned short* Abf, int strideA,
                 const unsigned short* Bpk, const float* bias, int Ktiles,
                 float* f32out, int ldF, unsigned short* bfout) {
  int lane = threadIdx.x & 31;
  int wave = threadIdx.x >> 5;          // 0..7
  int hi   = lane >> 4;
  int lcol = lane & 15;
  int nt0  = wave * 4;

  CFrag c[4];
#pragma unroll
  for (int j = 0; j < 4; ++j) {
    float bv = bias[(nt0 + j) * 16 + lcol];
#pragma unroll
    for (int r = 0; r < 8; ++r) c[j].f[r] = bv;
  }

  for (int kt = 0; kt < Ktiles; ++kt) {
    ABfrag a = load_a_frag(Abf, strideA, kt, lcol, hi);
#pragma unroll
    for (int j = 0; j < 4; ++j) {
      size_t tidx = (size_t)(nt0 + j) * Ktiles + kt;
      ABfrag b = load_b_frag(Bpk, tidx, lane);
      __builtin_prefetch((const char*)Bpk + ((tidx + 1) << 10), 0, 3);
      c[j].v = wmma_bf16(a, b, c[j].v);
    }
  }

#pragma unroll
  for (int j = 0; j < 4; ++j) {
    int n = (nt0 + j) * 16 + lcol;
#pragma unroll
    for (int r = 0; r < 8; ++r)
      ode_epilogue<MODE>(c[j].f[r], r + 8 * hi, n, f32out, ldF, bfout);
  }
}

// Generic n-tile count (used for W_out, Ntiles=22).
template <int MODE>
__device__ __forceinline__
void wg_matmul(const unsigned short* Abf, int strideA,
               const unsigned short* Bpk, const float* bias,
               int Ktiles, int Ntiles,
               float* f32out, int ldF, unsigned short* bfout) {
  int lane = threadIdx.x & 31;
  int wave = threadIdx.x >> 5;
  int nwaves = blockDim.x >> 5;
  int hi   = lane >> 4;
  int lcol = lane & 15;
  for (int nt = wave; nt < Ntiles; nt += nwaves) {
    CFrag c;
    float bv = bias[nt * 16 + lcol];
#pragma unroll
    for (int r = 0; r < 8; ++r) c.f[r] = bv;
    size_t tbase = (size_t)nt * Ktiles;
    for (int kt = 0; kt < Ktiles; ++kt) {
      ABfrag a = load_a_frag(Abf, strideA, kt, lcol, hi);
      ABfrag b = load_b_frag(Bpk, tbase + kt, lane);
      __builtin_prefetch((const char*)Bpk + ((tbase + kt + 1) << 10), 0, 3);
      c.v = wmma_bf16(a, b, c.v);
    }
    int n = nt * 16 + lcol;
#pragma unroll
    for (int r = 0; r < 8; ++r)
      ode_epilogue<MODE>(c.f[r], r + 8 * hi, n, f32out, ldF, bfout);
  }
}

// ---------------- the big one: full RK4 scan, 16 batch rows per workgroup ---
__global__ __launch_bounds__(256)
void ode_kernel(const float* __restrict__ times,
                const float* __restrict__ s0g,
                const unsigned short* __restrict__ pk_win,
                const float* __restrict__ ode_bin,
                const unsigned short* __restrict__ pk_rb1,
                const float* __restrict__ rb_b1,
                const unsigned short* __restrict__ pk_rb2,
                const float* __restrict__ rb_b2,
                const unsigned short* __restrict__ pk_wout,
                const float* __restrict__ ode_bout,
                unsigned short* __restrict__ predh_bf,
                float* __restrict__ loc_embed_out) {
  __shared__ __align__(16) float sS[16 * STATE];
  __shared__ __align__(16) float sTmp[16 * STATE];
  __shared__ __align__(16) float sAcc[16 * STATE];
  __shared__ __align__(16) float sK[16 * STATE];
  __shared__ __align__(16) float sU[16 * 512];               // residual u, f32
  __shared__ __align__(16) unsigned short sUbf[16 * 512];    // u as bf16
  __shared__ __align__(16) unsigned short sXbf[16 * 512];    // x / t scratch, bf16

  const int tid = threadIdx.x;
  const int m0  = blockIdx.x * 16;
  const float c24 = 6.2831853071795864769f / 24.0f;

  // load initial state
  for (int idx = tid; idx < 16 * STATE; idx += 256)
    sS[idx] = s0g[(size_t)(m0 + idx / STATE) * STATE + (idx % STATE)];
  __syncthreads();

  for (int step = 0; step < TT - 1; ++step) {
    float t0 = times[step], t1 = times[step + 1];
    float dt = t1 - t0;

    for (int stage = 0; stage < 4; ++stage) {
      float tcur = (stage == 0) ? t0 : (stage == 3 ? t1 : t0 + 0.5f * dt);
      const float* in = (stage == 0) ? sS : sTmp;
      float sv = __sinf(tcur * c24);
      float cv = __cosf(tcur * c24);

      // build x = [s, sin, cos, 0-pad] as bf16, stride 512, Kpad=384
      for (int idx = tid; idx < 16 * 384; idx += 256) {
        int m = idx / 384, cc = idx % 384;
        float v = (cc < STATE) ? in[m * STATE + cc]
                               : (cc == STATE ? sv : (cc == STATE + 1 ? cv : 0.0f));
        sXbf[m * 512 + cc] = f32_to_bf16_bits(v);
      }
      __syncthreads();

      // u = relu(x @ W_in + b_in)
      wg_matmul32<0>(sXbf, 512, pk_win, ode_bin, 12, sU, 512, sUbf);
      __syncthreads();

      // residual blocks
      for (int i = 0; i < NRB; ++i) {
        wg_matmul32<1>(sUbf, 512, pk_rb1 + (size_t)i * (512 * 512),
                       rb_b1 + i * 512, 16, nullptr, 0, sXbf);
        __syncthreads();
        wg_matmul32<2>(sXbf, 512, pk_rb2 + (size_t)i * (512 * 512),
                       rb_b2 + i * 512, 16, sU, 512, sUbf);
        __syncthreads();
      }

      // k = u @ W_out + b_out
      wg_matmul<3>(sUbf, 512, pk_wout, ode_bout, 16, 22, sK, STATE, nullptr);
      __syncthreads();

      // RK4 combination
      if (stage < 3) {
        float cin  = (stage == 0) ? 1.0f : 2.0f;           // weight of k in acc
        float cstp = (stage == 2) ? dt : 0.5f * dt;         // step for next probe
        for (int idx = tid; idx < 16 * STATE; idx += 256) {
          float k = sK[idx];
          if (stage == 0) sAcc[idx] = k; else sAcc[idx] += cin * k;
          sTmp[idx] = sS[idx] + cstp * k;
        }
        __syncthreads();
      } else {
        float d6 = dt / 6.0f;
        for (int idx = tid; idx < 16 * STATE; idx += 256) {
          float snew = sS[idx] + d6 * (sAcc[idx] + sK[idx]);
          sS[idx] = snew;
          int m = idx / STATE, cc = idx % STATE;
          size_t row = (size_t)(m0 + m) * TT + (step + 1);
          if (cc < HID) {
            predh_bf[row * HID + cc] = f32_to_bf16_bits(snew);
          } else if (cc < HID + ZED) {
            loc_embed_out[row * ZED + (cc - HID)] = snew;
          }
        }
        __syncthreads();
      }
    }
  }
}

// ---------------- small elementwise kernels ---------------------------------
__global__ void enc_build_kernel(const float* __restrict__ pf,
                                 const int* __restrict__ home,
                                 const int* __restrict__ work,
                                 const float* __restrict__ zone_emb,
                                 unsigned short* __restrict__ encA) {
  int idx = blockIdx.x * blockDim.x + threadIdx.x;
  if (idx >= BATCH * 160) return;
  int b = idx / 160, c = idx % 160;
  float v;
  if (c < PF)            v = pf[b * PF + c];
  else if (c < PF + ZED) v = zone_emb[(size_t)home[b] * ZED + (c - PF)];
  else if (c < PF + 2 * ZED) v = zone_emb[(size_t)work[b] * ZED + (c - PF - ZED)];
  else v = 0.0f;
  encA[idx] = f32_to_bf16_bits(v);
}

__global__ void s0_kernel(const float* __restrict__ hp,      // (B,512) raw enc out
                          const float* __restrict__ eps,
                          const int* __restrict__ home,
                          const int* __restrict__ purp,
                          const float* __restrict__ zone_emb,
                          const float* __restrict__ purp_emb,
                          float* __restrict__ s0g,
                          unsigned short* __restrict__ predh_bf,
                          float* __restrict__ loc_embed_out,
                          float* __restrict__ mu_out,
                          float* __restrict__ lv_out) {
  int idx = blockIdx.x * blockDim.x + threadIdx.x;
  if (idx >= BATCH * STATE) return;
  int b = idx / STATE, c = idx % STATE;
  if (c < HID) {
    float mu = hp[(size_t)b * 512 + c];
    float lv = hp[(size_t)b * 512 + HID + c];
    float h0 = mu + expf(0.5f * lv) * eps[(size_t)b * HID + c];
    mu_out[(size_t)b * HID + c] = mu;
    lv_out[(size_t)b * HID + c] = lv;
    s0g[idx] = h0;
    predh_bf[((size_t)b * TT) * HID + c] = f32_to_bf16_bits(h0);
  } else if (c < HID + ZED) {
    int j = c - HID;
    float he = zone_emb[(size_t)home[b] * ZED + j];
    s0g[idx] = he;
    loc_embed_out[((size_t)b * TT) * ZED + j] = he;
  } else {
    int j = c - HID - ZED;
    s0g[idx] = purp_emb[(size_t)purp[b] * PED + j];
  }
}

// ---------------- host side --------------------------------------------------
extern "C" void kernel_launch(void* const* d_in, const int* in_sizes, int n_in,
                              void* d_out, int out_size, void* d_ws, size_t ws_size,
                              hipStream_t stream) {
  const float* person_features = (const float*)d_in[0];
  const int*   home_zone_id    = (const int*)d_in[1];
  const int*   work_zone_id    = (const int*)d_in[2];
  const int*   start_purp_id   = (const int*)d_in[3];
  const float* times           = (const float*)d_in[4];
  const float* eps             = (const float*)d_in[5];
  const float* zone_emb        = (const float*)d_in[6];
  const float* purp_emb        = (const float*)d_in[7];
  const float* enc_w1          = (const float*)d_in[8];
  const float* enc_b1          = (const float*)d_in[9];
  const float* enc_w2          = (const float*)d_in[10];
  const float* enc_b2          = (const float*)d_in[11];
  const float* ode_win         = (const float*)d_in[12];
  const float* ode_bin         = (const float*)d_in[13];
  const float* rb_w1           = (const float*)d_in[14];
  const float* rb_b1           = (const float*)d_in[15];
  const float* rb_w2           = (const float*)d_in[16];
  const float* rb_b2           = (const float*)d_in[17];
  const float* ode_wout        = (const float*)d_in[18];
  const float* ode_bout        = (const float*)d_in[19];
  const float* dec_loc_w       = (const float*)d_in[20];
  const float* dec_loc_b       = (const float*)d_in[21];
  const float* dec_purp_w      = (const float*)d_in[22];
  const float* dec_purp_b      = (const float*)d_in[23];
  (void)in_sizes; (void)n_in; (void)out_size; (void)ws_size;

  // d_out layout (f32): loc_logits | loc_embed | purp_logits | mu | log_var
  float* out_loc   = (float*)d_out;
  float* out_emb   = out_loc + (size_t)BATCH * TT * NZ;
  float* out_purp  = out_emb + (size_t)BATCH * TT * ZED;
  float* out_mu    = out_purp + (size_t)BATCH * TT * NP;
  float* out_lv    = out_mu  + (size_t)BATCH * HID;

  // workspace carve-up (256B aligned)
  size_t off = 0;
  auto alloc = [&](size_t bytes) -> char* {
    char* p = (char*)d_ws + off;
    off += (bytes + 255) & ~(size_t)255;
    return p;
  };
  const int KT_ENC1 = 5,  NT_ENC1 = 32;   // 144(->160) x 512
  const int KT_512  = 16, NT_512  = 32;   // 512 x 512
  const int KT_WIN  = 12, NT_WIN  = 32;   // 354(->384) x 512
  const int KT_WOUT = 16, NT_WOUT = 22;   // 512 x 352
  const int KT_LOC  = 8,  NT_LOC  = 125;  // 256 x 2000
  const int KT_PUR  = 8,  NT_PUR  = 1;    // 256 x 8(->16)

  unsigned short* pk_encw1 = (unsigned short*)alloc((size_t)KT_ENC1 * NT_ENC1 * 512 * 2);
  unsigned short* pk_encw2 = (unsigned short*)alloc((size_t)KT_512 * NT_512 * 512 * 2);
  unsigned short* pk_win   = (unsigned short*)alloc((size_t)KT_WIN * NT_WIN * 512 * 2);
  unsigned short* pk_rb1   = (unsigned short*)alloc((size_t)NRB * KT_512 * NT_512 * 512 * 2);
  unsigned short* pk_rb2   = (unsigned short*)alloc((size_t)NRB * KT_512 * NT_512 * 512 * 2);
  unsigned short* pk_wout  = (unsigned short*)alloc((size_t)KT_WOUT * NT_WOUT * 512 * 2);
  unsigned short* pk_loc   = (unsigned short*)alloc((size_t)KT_LOC * NT_LOC * 512 * 2);
  unsigned short* pk_purp  = (unsigned short*)alloc((size_t)KT_PUR * NT_PUR * 512 * 2);
  unsigned short* encA     = (unsigned short*)alloc((size_t)BATCH * 160 * 2);
  unsigned short* hbf      = (unsigned short*)alloc((size_t)BATCH * 512 * 2);
  float*          hp       = (float*)alloc((size_t)BATCH * 512 * 4);
  float*          s0g      = (float*)alloc((size_t)BATCH * STATE * 4);
  unsigned short* predh    = (unsigned short*)alloc((size_t)BATCH * TT * HID * 2);

  // ---- pack all weights into WMMA B-fragment order ----
  pack_b_kernel<<<KT_ENC1 * NT_ENC1, 32, 0, stream>>>(enc_w1, pk_encw1, PF + 2 * ZED, ENC, KT_ENC1, NT_ENC1);
  pack_b_kernel<<<KT_512 * NT_512, 32, 0, stream>>>(enc_w2, pk_encw2, ENC, 2 * HID, KT_512, NT_512);
  pack_b_kernel<<<KT_WIN * NT_WIN, 32, 0, stream>>>(ode_win, pk_win, STATE + 2, OHID, KT_WIN, NT_WIN);
  for (int i = 0; i < NRB; ++i) {
    pack_b_kernel<<<KT_512 * NT_512, 32, 0, stream>>>(rb_w1 + (size_t)i * OHID * OHID,
        pk_rb1 + (size_t)i * KT_512 * NT_512 * 512, OHID, OHID, KT_512, NT_512);
    pack_b_kernel<<<KT_512 * NT_512, 32, 0, stream>>>(rb_w2 + (size_t)i * OHID * OHID,
        pk_rb2 + (size_t)i * KT_512 * NT_512 * 512, OHID, OHID, KT_512, NT_512);
  }
  pack_b_kernel<<<KT_WOUT * NT_WOUT, 32, 0, stream>>>(ode_wout, pk_wout, OHID, STATE, KT_WOUT, NT_WOUT);
  pack_b_kernel<<<KT_LOC * NT_LOC, 32, 0, stream>>>(dec_loc_w, pk_loc, HID, NZ, KT_LOC, NT_LOC);
  pack_b_kernel<<<KT_PUR * NT_PUR, 32, 0, stream>>>(dec_purp_w, pk_purp, HID, NP, KT_PUR, NT_PUR);

  // ---- encoder ----
  enc_build_kernel<<<(BATCH * 160 + 255) / 256, 256, 0, stream>>>(
      person_features, home_zone_id, work_zone_id, zone_emb, encA);
  gemm_kernel<<<dim3(BATCH / 16, 8), 128, 0, stream>>>(
      encA, 160, pk_encw1, enc_b1, KT_ENC1, NT_ENC1, /*relu*/1,
      nullptr, hbf, 512, 512);
  gemm_kernel<<<dim3(BATCH / 16, 8), 128, 0, stream>>>(
      hbf, 512, pk_encw2, enc_b2, KT_512, NT_512, /*none*/0,
      hp, nullptr, 512, 512);
  s0_kernel<<<(BATCH * STATE + 255) / 256, 256, 0, stream>>>(
      hp, eps, home_zone_id, start_purp_id, zone_emb, purp_emb,
      s0g, predh, out_emb, out_mu, out_lv);

  // ---- full RK4 scan, one workgroup per 16 batch rows ----
  ode_kernel<<<BATCH / 16, 256, 0, stream>>>(
      times, s0g, pk_win, ode_bin, pk_rb1, rb_b1, pk_rb2, rb_b2,
      pk_wout, ode_bout, predh, out_emb);

  // ---- decoders over (B*T, HID) ----
  gemm_kernel<<<dim3(BATCH * TT / 16, (NT_LOC + 3) / 4), 128, 0, stream>>>(
      predh, HID, pk_loc, dec_loc_b, KT_LOC, NT_LOC, 0,
      out_loc, nullptr, NZ, NZ);
  gemm_kernel<<<dim3(BATCH * TT / 16, 1), 128, 0, stream>>>(
      predh, HID, pk_purp, dec_purp_b, KT_PUR, NT_PUR, 0,
      out_purp, nullptr, NP, NP);
}
